// TemporalGraphResidualMLP_19842748907735
// MI455X (gfx1250) — compile-verified, matrix operand order checked by reading, not run on hardware
//
#include <hip/hip_runtime.h>

// ---------------------------------------------------------------------------
// Types for CDNA5 WMMA (wave32, v_wmma_f32_16x16x32_bf16)
// ---------------------------------------------------------------------------
typedef __bf16 bf16;
typedef bf16  v16bf __attribute__((ext_vector_type(16)));
typedef bf16  v8bf  __attribute__((ext_vector_type(8)));
typedef float v8f   __attribute__((ext_vector_type(8)));

#define BATCH 8
#define NN    2000
#define LL    20
#define DIN   128
#define DH    256
#define BOT   64
#define RANKD 16
#define DFF   512
#define BN_TOT (BATCH * NN)      // 16000

// Input tensor order (setup_inputs dict, params flattened in insertion order)
enum {
  IN_X = 0, IN_MASK, IN_HIST,
  ENC_LN_S, ENC_LN_B, ENC_DOWN_W, ENC_DOWN_B, ENC_UP_W, ENC_UP_B,
  FEAT_W, FEAT_B, TPROJ_W, TPROJ_B, TPOS,
  ATTN_LN_S, ATTN_LN_B,
  WQ_W, WQ_B, WK_W, WK_B, WV_W, WV_B, WO_W, WO_B,
  FF_LN_S, FF_LN_B, FF1_W, FF1_B, FF2_W, FF2_B,
  GATE_LN_S, GATE_LN_B, G1_W, G1_B, G2_W, G2_B,
  POST_LN_S, POST_LN_B,
  B0_LN_S, B0_LN_B, B0_W1_W, B0_W1_B, B0_W2_W, B0_W2_B,
  B1_LN_S, B1_LN_B, B1_W1_W, B1_W1_B, B1_W2_W, B1_W2_B,
  FDG_WR, FDG_CORE, MIX_LOGIT,
  GB_LN_S, GB_LN_B, GB_W1_W, GB_W1_B, GB_W2_W, GB_W2_B,
  HEAD_LN_S, HEAD_LN_B, HEAD_W, HEAD_B
};

__device__ inline float gelu_f(float x) {
  return 0.5f * x * (1.0f + erff(x * 0.70710678118654752f));
}

// ---------------------------------------------------------------------------
// WMMA fragment loaders (ISA 7.12.2 16-bit layouts, wave32)
// ---------------------------------------------------------------------------
__device__ inline v16bf frag_a(const bf16* A, int lda, int m0, int kb, int lane) {
  const bf16* p = A + (size_t)(m0 + (lane & 15)) * lda + kb + ((lane >> 4) << 3);
  v8bf lo = *(const v8bf*)(p);        // ds_load_b128
  v8bf hi = *(const v8bf*)(p + 16);   // ds_load_b128
  return __builtin_shufflevector(lo, hi, 0, 1, 2, 3, 4, 5, 6, 7,
                                 8, 9, 10, 11, 12, 13, 14, 15);
}

// B fragment from pre-swizzled weights: per (kt,nt) tile, lane's 16 values are
// contiguous (32 bytes, 32B aligned) -> two global_load_b128.
__device__ inline v16bf frag_b_swz(const bf16* Wswz, int tiles_k, int kt, int nt,
                                   int lane) {
  const bf16* p = Wswz + ((((size_t)nt * tiles_k + kt) * 32 + lane) << 4);
  return *(const v16bf*)p;
}

// ---------------------------------------------------------------------------
// Block-cooperative WMMA GEMM for M <= 32 (staging buffers padded to 32 rows;
// garbage in pad rows only produces garbage in pad output rows, never read).
// Each wave owns a 16-wide column strip with TWO accumulators (rows 0-15 and
// 16-31), reusing every B fragment for two WMMAs. Stores are unconditional.
// ---------------------------------------------------------------------------
__device__ inline void gemm_bf16out(const bf16* A, int lda,
                                    const bf16* __restrict__ W,
                                    const float* __restrict__ bias,
                                    int K, int N, bf16* out, int ldo, int act) {
  int lane = threadIdx.x & 31, wave = (int)threadIdx.x >> 5;
  int nw = (int)blockDim.x >> 5;
  int Nt = N >> 4, Kt = K >> 5;
  for (int nt = wave; nt < Nt; nt += nw) {
    v8f acc0 = {0.f, 0.f, 0.f, 0.f, 0.f, 0.f, 0.f, 0.f};
    v8f acc1 = {0.f, 0.f, 0.f, 0.f, 0.f, 0.f, 0.f, 0.f};
    for (int kt = 0; kt < Kt; ++kt) {
      v16bf b  = frag_b_swz(W, Kt, kt, nt, lane);
      v16bf a0 = frag_a(A, lda, 0, kt << 5, lane);
      acc0 = __builtin_amdgcn_wmma_f32_16x16x32_bf16(false, a0, false, b,
                                                     (short)0, acc0, false, false);
      v16bf a1 = frag_a(A, lda, 16, kt << 5, lane);
      acc1 = __builtin_amdgcn_wmma_f32_16x16x32_bf16(false, a1, false, b,
                                                     (short)0, acc1, false, false);
    }
    int n  = (nt << 4) + (lane & 15);
    int mb = (lane >> 4) << 3;
    float bb = bias[n];
#pragma unroll
    for (int r = 0; r < 8; ++r) {
      float v0 = acc0[r] + bb;
      float v1 = acc1[r] + bb;
      if (act) { v0 = gelu_f(v0); v1 = gelu_f(v1); }
      out[(size_t)(mb + r) * ldo + n]      = (bf16)v0;
      out[(size_t)(16 + mb + r) * ldo + n] = (bf16)v1;
    }
  }
}

__device__ inline void gemm_f32out(const bf16* A, int lda,
                                   const bf16* __restrict__ W,
                                   const float* __restrict__ bias,
                                   int K, int N, float* out, int ldo, int accum) {
  int lane = threadIdx.x & 31, wave = (int)threadIdx.x >> 5;
  int nw = (int)blockDim.x >> 5;
  int Nt = N >> 4, Kt = K >> 5;
  for (int nt = wave; nt < Nt; nt += nw) {
    v8f acc0 = {0.f, 0.f, 0.f, 0.f, 0.f, 0.f, 0.f, 0.f};
    v8f acc1 = {0.f, 0.f, 0.f, 0.f, 0.f, 0.f, 0.f, 0.f};
    for (int kt = 0; kt < Kt; ++kt) {
      v16bf b  = frag_b_swz(W, Kt, kt, nt, lane);
      v16bf a0 = frag_a(A, lda, 0, kt << 5, lane);
      acc0 = __builtin_amdgcn_wmma_f32_16x16x32_bf16(false, a0, false, b,
                                                     (short)0, acc0, false, false);
      v16bf a1 = frag_a(A, lda, 16, kt << 5, lane);
      acc1 = __builtin_amdgcn_wmma_f32_16x16x32_bf16(false, a1, false, b,
                                                     (short)0, acc1, false, false);
    }
    int n  = (nt << 4) + (lane & 15);
    int mb = (lane >> 4) << 3;
    float bb = bias[n];
#pragma unroll
    for (int r = 0; r < 8; ++r) {
      float v0 = acc0[r] + bb;
      float v1 = acc1[r] + bb;
      size_t g0 = (size_t)(mb + r) * ldo + n;
      size_t g1 = (size_t)(16 + mb + r) * ldo + n;
      if (accum) { out[g0] += v0; out[g1] += v1; }
      else       { out[g0] = v0;  out[g1] = v1; }
    }
  }
}

// Wave-per-row LayerNorm: src fp32 (global or LDS) -> dst bf16
__device__ inline void wave_ln(const float* src, int lds_, int rows, int cols,
                               const float* s, const float* b, bf16* dst, int ldd) {
  int lane = threadIdx.x & 31, wave = (int)threadIdx.x >> 5;
  int nw = (int)blockDim.x >> 5;
  float invc = 1.0f / (float)cols;
  for (int r = wave; r < rows; r += nw) {
    const float* row = src + (size_t)r * lds_;
    float sum = 0.f, sq = 0.f;
    for (int c = lane; c < cols; c += 32) { float v = row[c]; sum += v; sq += v * v; }
#pragma unroll
    for (int o = 16; o > 0; o >>= 1) {
      sum += __shfl_xor(sum, o, 32);
      sq  += __shfl_xor(sq, o, 32);
    }
    float mean = sum * invc;
    float var  = sq * invc - mean * mean;
    float inv  = rsqrtf(var + 1e-5f);
    for (int c = lane; c < cols; c += 32)
      dst[(size_t)r * ldd + c] = (bf16)((row[c] - mean) * inv * s[c] + b[c]);
  }
}

// ---------------------------------------------------------------------------
// Shuffle-based block reductions (blockDim.x == 256 -> 8 waves).
// 2 barriers per reduction; cross-wave combine via 8 LDS partials.
// ---------------------------------------------------------------------------
__device__ inline float wave_sum(float v) {
#pragma unroll
  for (int o = 16; o > 0; o >>= 1) v += __shfl_xor(v, o, 32);
  return v;
}
__device__ inline float wave_max(float v) {
#pragma unroll
  for (int o = 16; o > 0; o >>= 1) v = fmaxf(v, __shfl_xor(v, o, 32));
  return v;
}
__device__ inline void wave_argmax(float& v, int& i) {
#pragma unroll
  for (int o = 16; o > 0; o >>= 1) {
    float ov = __shfl_xor(v, o, 32);
    int   oi = __shfl_xor(i, o, 32);
    if (ov > v || (ov == v && oi < i)) { v = ov; i = oi; }
  }
}
__device__ inline float blk_sum(float v, float* red8) {
  int tid = threadIdx.x, lane = tid & 31, w = tid >> 5;
  v = wave_sum(v);
  __syncthreads();
  if (lane == 0) red8[w] = v;
  __syncthreads();
  float x = (lane < 8) ? red8[lane] : 0.f;
  return wave_sum(x);
}
__device__ inline float blk_max(float v, float* red8) {
  int tid = threadIdx.x, lane = tid & 31, w = tid >> 5;
  v = wave_max(v);
  __syncthreads();
  if (lane == 0) red8[w] = v;
  __syncthreads();
  float x = (lane < 8) ? red8[lane] : -1e30f;
  return wave_max(x);
}
__device__ inline void blk_argmax(float& v, int& idx, float* red8, int* redi8) {
  int tid = threadIdx.x, lane = tid & 31, w = tid >> 5;
  wave_argmax(v, idx);
  __syncthreads();
  if (lane == 0) { red8[w] = v; redi8[w] = idx; }
  __syncthreads();
  float fv = (lane < 8) ? red8[lane] : -1e30f;
  int   fi = (lane < 8) ? redi8[lane] : 0x7fffffff;
  wave_argmax(fv, fi);
  v = fv; idx = fi;
}

// kth-largest value over a[0..NN) (LDS), k iterations of block argmax with
// register-resident candidates: each thread owns elements m = tid + 256*i.
// Removal is a local cndmask chain -- no LDS scratch, no extra barriers.
__device__ inline float topk_thresh(const float* a, int k, float* red8,
                                    int* redi8) {
  int d = threadIdx.x;
  float loc[8];
#pragma unroll
  for (int i = 0; i < 8; ++i) {
    int m = d + (i << 8);
    loc[i] = (m < NN) ? a[m] : -1e30f;
  }
  float thr = -1e30f;
  for (int it = 0; it < k; ++it) {
    float lv = -1e30f; int li = 0x7fffffff;
#pragma unroll
    for (int i = 0; i < 8; ++i) {
      int m = d + (i << 8);
      if (loc[i] > lv) { lv = loc[i]; li = m; }
    }
    float gv = lv; int gi = li;
    blk_argmax(gv, gi, red8, redi8);
    thr = gv;
#pragma unroll
    for (int i = 0; i < 8; ++i)
      if (gi == d + (i << 8)) loc[i] = -1e30f;   // owner removes (unique mod 256)
  }
  return thr;
}

// ---------------------------------------------------------------------------
// fp32 -> bf16 conversion with WMMA-B fragment swizzle (weights, once/call).
// ---------------------------------------------------------------------------
__global__ void k_cvt_swz(const float* __restrict__ src, bf16* __restrict__ dst,
                          int K, int N) {
  int i = blockIdx.x * blockDim.x + threadIdx.x;
  if (i >= K * N) return;
  int tiles_k = K >> 5;
  int e    = i & 15;
  int lane = (i >> 4) & 31;
  int t    = i >> 9;
  int kt   = t % tiles_k;
  int nt   = t / tiles_k;
  int j    = e >> 1, odd = e & 1;
  int n = (nt << 4) + (lane & 15);
  int k = (kt << 5) + ((lane >> 4) << 3) + ((j & 4) << 2) + ((j & 3) << 1) + odd;
  dst[i] = (bf16)src[(size_t)k * N + n];
}

// ---------------------------------------------------------------------------
// Encoder: x = (X + up(gelu(down(ln(X)))))*mf ; feat = x @ Wf + bf
// ---------------------------------------------------------------------------
__global__ __launch_bounds__(256) void k_encoder(
    const float* __restrict__ X, const unsigned char* __restrict__ mask,
    const float* ln_s, const float* ln_b,
    const bf16* wd, const float* bd, const bf16* wu, const float* bu,
    const bf16* wf, const float* bf_, float* __restrict__ feat) {
  __shared__ __align__(16) bf16 xbf[32 * DIN];
  __shared__ __align__(16) bf16 t1[32 * BOT];
  __shared__ float xf[32 * DIN];
  int tid = threadIdx.x;
  int r0 = blockIdx.x * 32;
  for (int i = tid; i < 32 * DIN; i += 256) xf[i] = X[(size_t)r0 * DIN + i];
  wave_ln(X + (size_t)r0 * DIN, DIN, 32, DIN, ln_s, ln_b, xbf, DIN);
  __syncthreads();
  gemm_bf16out(xbf, DIN, wd, bd, DIN, BOT, t1, BOT, 1);   // gelu
  __syncthreads();
  gemm_f32out(t1, BOT, wu, bu, BOT, DIN, xf, DIN, 1);     // xf = X + up(..)
  __syncthreads();
  for (int i = tid; i < 32 * DIN; i += 256) {
    int r = i >> 7;
    float mf = mask[r0 + r] ? 1.f : 0.f;
    xbf[i] = (bf16)(xf[i] * mf);
  }
  __syncthreads();
  gemm_f32out(xbf, DIN, wf, bf_, DIN, DH, feat + (size_t)r0 * DH, DH, 0);
}

// ---------------------------------------------------------------------------
// Fused temporal transformer: one block per node (B*N blocks), all in LDS.
// All 2-D LDS buffers are 32 rows so WMMA loads/stores are branchless; only
// rows < 20 are ever consumed.
// ---------------------------------------------------------------------------
__global__ __launch_bounds__(256) void k_temporal(
    const float* __restrict__ hist, const unsigned char* __restrict__ mask,
    const float* pw, const float* pb, const float* pos,
    const float* aln_s, const float* aln_b,
    const bf16* wq, const float* bq, const bf16* wk, const float* bk,
    const bf16* wv, const float* bv, const bf16* wo, const float* bo,
    const float* fln_s, const float* fln_b,
    const bf16* w1, const float* b1, const bf16* w2, const float* b2,
    float* __restrict__ temporal) {
  extern __shared__ char smem[];
  float* seq = (float*)smem;                   // 32*256 f32   (32768 B)
  bf16*  ybf = (bf16*)(seq + 32 * DH);         // 32*256 bf16  (16384 B)
  bf16*  qs  = ybf + 32 * DH;                  // 32*256 bf16 x3 (49152 B)
  bf16*  ks  = qs + 32 * DH;
  bf16*  vs  = ks + 32 * DH;
  float* att = (float*)(vs + 32 * DH);         // 4*20*20 f32  (6400 B)
  bf16*  os  = (bf16*)(att + 4 * LL * LL);     // 32*256 bf16  (16384 B)
  bf16*  mid = qs;                             // reuse qkv region: 32*512 bf16
  int tid = threadIdx.x;
  int bn = blockIdx.x;

  // 1. seq = proj(history) + pos
  for (int i = tid; i < LL * DH; i += 256) {
    int l = i >> 8, d = i & 255;
    seq[i] = hist[(size_t)bn * LL + l] * pw[d] + pb[d] + pos[i];
  }
  __syncthreads();
  // 2. pre-LN
  wave_ln(seq, DH, LL, DH, aln_s, aln_b, ybf, DH);
  __syncthreads();
  // 3. QKV projections (WMMA)
  gemm_bf16out(ybf, DH, wq, bq, DH, DH, qs, DH, 0);
  gemm_bf16out(ybf, DH, wk, bk, DH, DH, ks, DH, 0);
  gemm_bf16out(ybf, DH, wv, bv, DH, DH, vs, DH, 0);
  __syncthreads();
  // 4. attention scores + softmax (4 heads, 20x20 — VALU)
  for (int t = tid; t < 4 * LL * LL; t += 256) {
    int h = t / (LL * LL); int lm = t - h * LL * LL;
    int l = lm / LL, m = lm - l * LL;
    const bf16* qp = qs + l * DH + h * 64;
    const bf16* kp = ks + m * DH + h * 64;
    float sc = 0.f;
#pragma unroll 8
    for (int d = 0; d < 64; ++d) sc += (float)qp[d] * (float)kp[d];
    att[t] = sc * 0.125f;    // 1/sqrt(64)
  }
  __syncthreads();
  for (int r = tid; r < 4 * LL; r += 256) {
    float* a = att + r * LL;
    float mx = a[0];
    for (int m = 1; m < LL; ++m) mx = fmaxf(mx, a[m]);
    float sm = 0.f;
    for (int m = 0; m < LL; ++m) { float e = __expf(a[m] - mx); a[m] = e; sm += e; }
    float inv = 1.f / sm;
    for (int m = 0; m < LL; ++m) a[m] *= inv;
  }
  __syncthreads();
  // 5. o = att @ v
  for (int i = tid; i < LL * DH; i += 256) {
    int l = i >> 8, d = i & 255;
    int h = d >> 6;
    const float* a = att + (h * LL + l) * LL;
    float sc = 0.f;
#pragma unroll
    for (int m = 0; m < LL; ++m) sc += a[m] * (float)vs[m * DH + d];
    os[i] = (bf16)sc;
  }
  __syncthreads();
  // 6. seq += o @ wo + bo
  gemm_f32out(os, DH, wo, bo, DH, DH, seq, DH, 1);
  __syncthreads();
  // 7. FFN: seq += ff2(gelu(ff1(ln(seq))))
  wave_ln(seq, DH, LL, DH, fln_s, fln_b, ybf, DH);
  __syncthreads();
  gemm_bf16out(ybf, DH, w1, b1, DH, DFF, mid, DFF, 1);
  __syncthreads();
  gemm_f32out(mid, DFF, w2, b2, DFF, DH, seq, DH, 1);
  __syncthreads();
  // 8. temporal = last token * mask
  float mf = mask[bn] ? 1.f : 0.f;
  for (int d = tid; d < DH; d += 256)
    temporal[(size_t)bn * DH + d] = seq[(LL - 1) * DH + d] * mf;
}

// ---------------------------------------------------------------------------
// ctx = masked mean over nodes (temporal is already masked)
// ---------------------------------------------------------------------------
__global__ __launch_bounds__(256) void k_ctx(const float* __restrict__ temporal,
                                             const unsigned char* __restrict__ mask,
                                             float* __restrict__ ctx) {
  int b = blockIdx.x, d = threadIdx.x;
  float s = 0.f, c = 0.f;
  for (int n = 0; n < NN; ++n) {
    s += temporal[((size_t)b * NN + n) * DH + d];
    c += mask[b * NN + n] ? 1.f : 0.f;
  }
  ctx[b * DH + d] = s / fmaxf(c, 1.f);
}

// gate = sigmoid(g2(gelu(g1(ln(ctx)))))  (per batch, tiny)
__global__ __launch_bounds__(256) void k_gate(
    const float* __restrict__ ctx, const float* ln_s, const float* ln_b,
    const float* w1, const float* b1, const float* w2, const float* b2,
    float* __restrict__ gate) {
  __shared__ float t[DH], g1s[DH], red[8];
  int b = blockIdx.x, d = threadIdx.x;
  float v = ctx[b * DH + d];
  float mean = blk_sum(v, red) * (1.f / DH);
  float var  = blk_sum((v - mean) * (v - mean), red) * (1.f / DH);
  t[d] = (v - mean) * rsqrtf(var + 1e-5f) * ln_s[d] + ln_b[d];
  __syncthreads();
  float a = b1[d];
  for (int e = 0; e < DH; ++e) a += t[e] * w1[e * DH + d];
  g1s[d] = gelu_f(a);
  __syncthreads();
  float a2 = b2[d];
  for (int e = 0; e < DH; ++e) a2 += g1s[e] * w2[e * DH + d];
  gate[b * DH + d] = 1.f / (1.f + __expf(-a2));
}

// hidden = ln(feat + gate*temporal, post_ln) * mf
__global__ __launch_bounds__(256) void k_combine(
    const float* __restrict__ feat, const float* __restrict__ temporal,
    const float* __restrict__ gate, const unsigned char* __restrict__ mask,
    const float* ln_s, const float* ln_b, float* __restrict__ hidden) {
  __shared__ float red[8];
  int bn = blockIdx.x, d = threadIdx.x;
  int b = bn / NN;
  float v = feat[(size_t)bn * DH + d] + gate[b * DH + d] * temporal[(size_t)bn * DH + d];
  float mean = blk_sum(v, red) * (1.f / DH);
  float var  = blk_sum((v - mean) * (v - mean), red) * (1.f / DH);
  float mf = mask[bn] ? 1.f : 0.f;
  hidden[(size_t)bn * DH + d] =
      ((v - mean) * rsqrtf(var + 1e-5f) * ln_s[d] + ln_b[d]) * mf;
}

// ---------------------------------------------------------------------------
// Residual / graph MLP block: hidden = (hidden + w2(gelu(w1(A)+b1))+b2)*mf
// A = ln(src) if ln params given, else src directly. 32 rows per block.
// ---------------------------------------------------------------------------
__global__ __launch_bounds__(256) void k_mlp_block(
    const float* __restrict__ src, float* __restrict__ hidden,
    const unsigned char* __restrict__ mask,
    const float* ln_s, const float* ln_b,
    const bf16* w1, const float* b1, const bf16* w2, const float* b2, int N1) {
  extern __shared__ char smem[];
  bf16*  abf   = (bf16*)smem;                 // 32*256
  bf16*  mid   = abf + 32 * DH;               // 32*N1
  float* delta = (float*)(mid + 32 * N1);     // 32*256
  int tid = threadIdx.x;
  int r0 = blockIdx.x * 32;
  if (ln_s) {
    wave_ln(src + (size_t)r0 * DH, DH, 32, DH, ln_s, ln_b, abf, DH);
  } else {
    for (int i = tid; i < 32 * DH; i += 256)
      abf[i] = (bf16)src[(size_t)r0 * DH + i];
  }
  __syncthreads();
  gemm_bf16out(abf, DH, w1, b1, DH, N1, mid, N1, 1);   // gelu
  __syncthreads();
  gemm_f32out(mid, N1, w2, b2, N1, DH, delta, DH, 0);
  __syncthreads();
  for (int i = tid; i < 32 * DH; i += 256) {
    int r = i >> 8;
    float mf = mask[r0 + r] ? 1.f : 0.f;
    size_t g = (size_t)r0 * DH + i;
    hidden[g] = (hidden[g] + delta[i]) * mf;
  }
}

// R = hidden @ wr ; G = R @ core  (16 rows per block; tiny K=16 parts in VALU)
__global__ __launch_bounds__(256) void k_fdg(
    const float* __restrict__ hidden, const float* __restrict__ wr,
    const float* __restrict__ core, float* __restrict__ R, float* __restrict__ G) {
  __shared__ float Rs[16 * RANKD];
  int tid = threadIdx.x;
  int row = tid >> 4, j = tid & 15;
  size_t n = (size_t)blockIdx.x * 16 + row;
  float r = 0.f;
  for (int d = 0; d < DH; ++d) r += hidden[n * DH + d] * wr[d * RANKD + j];
  Rs[row * RANKD + j] = r;
  R[n * RANKD + j] = r;
  __syncthreads();
  float g = 0.f;
  for (int k = 0; k < RANKD; ++k) g += Rs[row * RANKD + k] * core[k * RANKD + j];
  G[n * RANKD + j] = g;
}

// Hn = normalized centered history rows
__global__ void k_hn(const float* __restrict__ hist, float* __restrict__ Hn) {
  int n = blockIdx.x * blockDim.x + threadIdx.x;
  if (n >= BN_TOT) return;
  const float* h = hist + (size_t)n * LL;
  float m = 0.f;
  for (int l = 0; l < LL; ++l) m += h[l];
  m *= (1.f / LL);
  float c[LL]; float nr = 0.f;
  for (int l = 0; l < LL; ++l) { c[l] = h[l] - m; nr += c[l] * c[l]; }
  float inv = 1.f / (sqrtf(nr) + 1e-6f);
  for (int l = 0; l < LL; ++l) Hn[(size_t)n * LL + l] = c[l] * inv;
}

// lnh = ln(hidden, gblock.ln) as bf16
__global__ __launch_bounds__(256) void k_lnh(
    const float* __restrict__ hidden, const float* ln_s, const float* ln_b,
    bf16* __restrict__ lnh) {
  __shared__ float red[8];
  int bn = blockIdx.x, d = threadIdx.x;
  float v = hidden[(size_t)bn * DH + d];
  float mean = blk_sum(v, red) * (1.f / DH);
  float var  = blk_sum((v - mean) * (v - mean), red) * (1.f / DH);
  lnh[(size_t)bn * DH + d] =
      (bf16)((v - mean) * rsqrtf(var + 1e-5f) * ln_s[d] + ln_b[d]);
}

// ---------------------------------------------------------------------------
// Graph row kernel: per (b,n) row build A_fdg softmax + A_roll topk + mix +
// final topk, then sparse-gather agg = sum_m A[n,m] * lnh[m].
// ---------------------------------------------------------------------------
__global__ __launch_bounds__(256) void k_graph(
    const float* __restrict__ R, const float* __restrict__ G,
    const float* __restrict__ Hn, const bf16* __restrict__ lnh,
    const unsigned char* __restrict__ mask, const float* __restrict__ mix_logit,
    float* __restrict__ agg) {
  __shared__ float sf[NN], cr[NN];
  __shared__ float gv[RANKD], hv[LL];
  __shared__ float red8[8];
  __shared__ int   redi8[8];
  __shared__ int   idxs[256];
  __shared__ float wts[256];
  __shared__ int   cnt;
  int bn = blockIdx.x;
  int b = bn / NN, n = bn - b * NN;
  int d = threadIdx.x;
  if (d < RANKD) gv[d] = G[(size_t)bn * RANKD + d];
  else if (d < RANKD + LL) hv[d - RANKD] = Hn[(size_t)bn * LL + (d - RANKD)];
  if (d == 0) cnt = 0;
  __syncthreads();
  float mfn = mask[bn] ? 1.f : 0.f;

  // FDG scores (masked) and rolling correlations
  for (int m = d; m < NN; m += 256) {
    size_t gm = (size_t)b * NN + m;
    bool pair = (mfn != 0.f) && mask[gm];
    float s = 0.f;
    const float* rp = R + gm * RANKD;
#pragma unroll
    for (int j = 0; j < RANKD; ++j) s += gv[j] * rp[j];
    sf[m] = pair ? s * 0.25f : -1e9f;   // 1/(tau*sqrt(rank)) = 0.25
    float c = 0.f;
    const float* hp = Hn + gm * LL;
#pragma unroll
    for (int l = 0; l < LL; ++l) c += hv[l] * hp[l];
    c = fmaxf(c, 0.f);
    cr[m] = (m == n) ? 0.f : c;
  }
  __syncthreads();

  // A_fdg = softmax(sf) * mfn
  float lmx = -1e30f;
  for (int m = d; m < NN; m += 256) lmx = fmaxf(lmx, sf[m]);
  float mx = blk_max(lmx, red8);
  float lsum = 0.f;
  for (int m = d; m < NN; m += 256) { float e = __expf(sf[m] - mx); sf[m] = e; lsum += e; }
  float ssum = blk_sum(lsum, red8);
  float sinv = mfn / fmaxf(ssum, 1e-30f);
  for (int m = d; m < NN; m += 256) sf[m] *= sinv;
  __syncthreads();

  // A_roll: topk(20) on cr, mask, normalize
  float thrC = topk_thresh(cr, 20, red8, redi8);
  float rs = 0.f;
  for (int m = d; m < NN; m += 256) {
    size_t gm = (size_t)b * NN + m;
    bool pair = (mfn != 0.f) && mask[gm];
    float a = (cr[m] >= thrC && pair) ? cr[m] : 0.f;
    cr[m] = a; rs += a;
  }
  float rsum = blk_sum(rs, red8);
  float rinv = 1.f / fmaxf(rsum, 1e-6f);

  // mix + row_normalize
  float mixv = 1.f / (1.f + __expf(-mix_logit[0]));
  float al = 0.f;
  for (int m = d; m < NN; m += 256) {
    size_t gm = (size_t)b * NN + m;
    bool pair = (mfn != 0.f) && mask[gm];
    float a = pair ? (mixv * sf[m] + (1.f - mixv) * cr[m] * rinv) : 0.f;
    sf[m] = a; al += a;
  }
  float asum = blk_sum(al, red8);
  float ainv = 1.f / fmaxf(asum, 1e-6f);
  for (int m = d; m < NN; m += 256) sf[m] *= ainv;
  __syncthreads();

  // final topk(20)
  float thr2 = topk_thresh(sf, 20, red8, redi8);
  float fl = 0.f;
  for (int m = d; m < NN; m += 256) {
    size_t gm = (size_t)b * NN + m;
    bool pair = (mfn != 0.f) && mask[gm];
    float a = (sf[m] >= thr2 && pair) ? sf[m] : 0.f;
    sf[m] = a; fl += a;
  }
  float fsum = blk_sum(fl, red8);
  float finv = 1.f / fmaxf(fsum, 1e-6f);

  // compact nonzeros (~20 entries) then sparse gather of lnh rows
  for (int m = d; m < NN; m += 256) {
    if (sf[m] > 0.f) {
      int p = atomicAdd(&cnt, 1);
      if (p < 256) { idxs[p] = m; wts[p] = sf[m] * finv; }
    }
  }
  __syncthreads();
  int nnz = cnt < 256 ? cnt : 256;
  float acc = 0.f;
  for (int i = 0; i < nnz; ++i)
    acc += wts[i] * (float)lnh[((size_t)b * NN + idxs[i]) * DH + d];
  agg[(size_t)bn * DH + d] = acc;
}

// y = head(gelu(ln(hidden))) masked
__global__ __launch_bounds__(256) void k_head(
    const float* __restrict__ hidden, const unsigned char* __restrict__ mask,
    const float* ln_s, const float* ln_b, const float* hw, const float* hb,
    float* __restrict__ out) {
  __shared__ float red[8];
  int bn = blockIdx.x, d = threadIdx.x;
  float v = hidden[(size_t)bn * DH + d];
  float mean = blk_sum(v, red) * (1.f / DH);
  float var  = blk_sum((v - mean) * (v - mean), red) * (1.f / DH);
  float t = gelu_f((v - mean) * rsqrtf(var + 1e-5f) * ln_s[d] + ln_b[d]);
  float y = blk_sum(t * hw[d], red) + hb[0];
  if (d == 0) out[bn] = mask[bn] ? y : 0.f;
}

// ---------------------------------------------------------------------------
// Host launcher
// ---------------------------------------------------------------------------
extern "C" void kernel_launch(void* const* d_in, const int* in_sizes, int n_in,
                              void* d_out, int out_size, void* d_ws, size_t ws_size,
                              hipStream_t stream) {
  (void)in_sizes; (void)n_in; (void)out_size; (void)ws_size;
  const float* X = (const float*)d_in[IN_X];
  const unsigned char* mask = (const unsigned char*)d_in[IN_MASK];
  const float* hist = (const float*)d_in[IN_HIST];
  auto F = [&](int i) { return (const float*)d_in[i]; };

  char* p = (char*)d_ws;
  auto alloc = [&](size_t bytes) -> char* {
    char* r = p;
    p += (bytes + 255) & ~(size_t)255;
    return r;
  };
  // convert + swizzle a [K,N] fp32 weight into WMMA-B fragment order
  auto cvt = [&](int idx, int K, int N) -> bf16* {
    size_t n = (size_t)K * N;
    bf16* dst = (bf16*)alloc(n * sizeof(bf16));
    k_cvt_swz<<<dim3((unsigned)((n + 255) / 256)), dim3(256), 0, stream>>>(
        F(idx), dst, K, N);
    return dst;
  };

  // Weight conversions to swizzled bf16 (once per call)
  bf16* wd   = cvt(ENC_DOWN_W, DIN, BOT);
  bf16* wu   = cvt(ENC_UP_W,   BOT, DIN);
  bf16* wf   = cvt(FEAT_W,     DIN, DH);
  bf16* wq   = cvt(WQ_W, DH, DH);
  bf16* wk   = cvt(WK_W, DH, DH);
  bf16* wv   = cvt(WV_W, DH, DH);
  bf16* wo   = cvt(WO_W, DH, DH);
  bf16* f1   = cvt(FF1_W, DH, DFF);
  bf16* f2   = cvt(FF2_W, DFF, DH);
  bf16* b0w1 = cvt(B0_W1_W, DH, DFF);
  bf16* b0w2 = cvt(B0_W2_W, DFF, DH);
  bf16* b1w1 = cvt(B1_W1_W, DH, DFF);
  bf16* b1w2 = cvt(B1_W2_W, DFF, DH);
  bf16* gw1  = cvt(GB_W1_W, DH, DH);
  bf16* gw2  = cvt(GB_W2_W, DH, DH);

  // fp32 activation buffers
  float* feat     = (float*)alloc((size_t)BN_TOT * DH * 4);
  float* temporal = (float*)alloc((size_t)BN_TOT * DH * 4);
  float* hidden   = (float*)alloc((size_t)BN_TOT * DH * 4);
  float* agg      = (float*)alloc((size_t)BN_TOT * DH * 4);
  float* Rb       = (float*)alloc((size_t)BN_TOT * RANKD * 4);
  float* Gb       = (float*)alloc((size_t)BN_TOT * RANKD * 4);
  float* Hb       = (float*)alloc((size_t)BN_TOT * LL * 4);
  float* ctx      = (float*)alloc((size_t)BATCH * DH * 4);
  float* gate     = (float*)alloc((size_t)BATCH * DH * 4);
  bf16*  lnh      = (bf16*)alloc((size_t)BN_TOT * DH * 2);

  // Stage 1: encoder + feature projection
  k_encoder<<<BN_TOT / 32, 256, 0, stream>>>(
      X, mask, F(ENC_LN_S), F(ENC_LN_B), wd, F(ENC_DOWN_B), wu, F(ENC_UP_B),
      wf, F(FEAT_B), feat);

  // Stage 1b: normalized history rows (independent)
  k_hn<<<(BN_TOT + 255) / 256, 256, 0, stream>>>(hist, Hb);

  // Stage 2: fused per-node temporal transformer (bulk of the FLOPs, WMMA)
  size_t smem_t = (size_t)32 * DH * 4        // seq f32 (padded)
                + (size_t)32 * DH * 2        // ybf
                + (size_t)3 * 32 * DH * 2    // q,k,v
                + (size_t)4 * LL * LL * 4    // att
                + (size_t)32 * DH * 2;       // o
  k_temporal<<<BN_TOT, 256, (unsigned)smem_t, stream>>>(
      hist, mask, F(TPROJ_W), F(TPROJ_B), F(TPOS), F(ATTN_LN_S), F(ATTN_LN_B),
      wq, F(WQ_B), wk, F(WK_B), wv, F(WV_B), wo, F(WO_B),
      F(FF_LN_S), F(FF_LN_B), f1, F(FF1_B), f2, F(FF2_B), temporal);

  // Stage 3: market gate
  k_ctx<<<BATCH, 256, 0, stream>>>(temporal, mask, ctx);
  k_gate<<<BATCH, 256, 0, stream>>>(ctx, F(GATE_LN_S), F(GATE_LN_B),
                                    F(G1_W), F(G1_B), F(G2_W), F(G2_B), gate);

  // Stage 4: combine + post LN
  k_combine<<<BN_TOT, 256, 0, stream>>>(feat, temporal, gate, mask,
                                        F(POST_LN_S), F(POST_LN_B), hidden);

  // Stage 5: residual MLP blocks (WMMA)
  size_t smem_b512 = 32 * DH * 2 + 32 * DFF * 2 + 32 * DH * 4;
  k_mlp_block<<<BN_TOT / 32, 256, (unsigned)smem_b512, stream>>>(
      hidden, hidden, mask, F(B0_LN_S), F(B0_LN_B),
      b0w1, F(B0_W1_B), b0w2, F(B0_W2_B), DFF);
  k_mlp_block<<<BN_TOT / 32, 256, (unsigned)smem_b512, stream>>>(
      hidden, hidden, mask, F(B1_LN_S), F(B1_LN_B),
      b1w1, F(B1_W1_B), b1w2, F(B1_W2_B), DFF);

  // Stage 6: FDG factors, graph-LN, sparse graph aggregation
  k_fdg<<<BN_TOT / 16, 256, 0, stream>>>(hidden, F(FDG_WR), F(FDG_CORE), Rb, Gb);
  k_lnh<<<BN_TOT, 256, 0, stream>>>(hidden, F(GB_LN_S), F(GB_LN_B), lnh);
  k_graph<<<BN_TOT, 256, 0, stream>>>(Rb, Gb, Hb, lnh, mask, F(MIX_LOGIT), agg);

  // Stage 7: graph message-passing MLP (no LN on agg; WMMA)
  size_t smem_b256 = 32 * DH * 2 + 32 * DH * 2 + 32 * DH * 4;
  k_mlp_block<<<BN_TOT / 32, 256, (unsigned)smem_b256, stream>>>(
      agg, hidden, mask, nullptr, nullptr,
      gw1, F(GB_W1_B), gw2, F(GB_W2_B), DH);

  // Stage 8: prediction head
  k_head<<<BN_TOT, 256, 0, stream>>>(hidden, mask, F(HEAD_LN_S), F(HEAD_LN_B),
                                     F(HEAD_W), F(HEAD_B), (float*)d_out);
}
